// SelfAttention_38268158607606
// MI455X (gfx1250) — matrix operand order repaired and checked by reading, not compiled, not run on hardware
//
#include <hip/hip_runtime.h>

typedef _Float16 h16;
typedef __attribute__((ext_vector_type(8)))  _Float16 v8h;
typedef __attribute__((ext_vector_type(16))) _Float16 v16h;
typedef __attribute__((ext_vector_type(8)))  float    v8f;
typedef unsigned int u32;
typedef __attribute__((ext_vector_type(4))) u32 u32x4;
typedef __attribute__((ext_vector_type(8))) int i32x8;
typedef __attribute__((ext_vector_type(4))) int i32x4;

#define C_DIM 256
#define N_DIM 256
#define XT_S  264   // padded row stride (halves / f32) for 256-wide buffers
#define QK_S  40    // padded row stride (halves) for 32-wide Q/K buffers
#define VS_BYTE_OFF (256 * XT_S * 2)  // byte offset of Vs region in LDS

#if defined(__gfx1250__) && __has_builtin(__builtin_amdgcn_tensor_load_to_lds) && \
    __has_builtin(__builtin_amdgcn_s_wait_tensorcnt)
#define USE_TDM 1
#else
#define USE_TDM 0
#endif

__device__ __forceinline__ v16h mk16(v8h lo, v8h hi) {
  v16h r;
#pragma unroll
  for (int i = 0; i < 8; ++i) { r[i] = lo[i]; r[i + 8] = hi[i]; }
  return r;
}

__device__ __forceinline__ v8f wmma_f16(v16h a, v16h b, v8f c) {
  return __builtin_amdgcn_wmma_f32_16x16x32_f16(false, a, false, b, (short)0, c,
                                                false, false);
}

#if USE_TDM
// TDM: load 16x16x128 f32 tile (strides 1,128,16384 elements) into LDS at
// lds_byte_off, padding +32B every 1KB so staged rows have stride 264 f32.
__device__ __forceinline__ void tdm_load_window_f32(unsigned long long gaddr,
                                                    u32 lds_byte_off) {
  u32x4 g0;
  g0[0] = 1u;                                   // count=1, user descriptor
  g0[1] = lds_byte_off;                         // lds_addr
  g0[2] = (u32)(gaddr & 0xffffffffu);           // global_addr[31:0]
  g0[3] = (u32)((gaddr >> 32) & 0x1ffffffu)     // global_addr[56:32]
          | (2u << 30);                         // type=2 ("image")
  i32x8 g1;
  g1[0] = (int)((2u << 16)      // data_size = 4B
                | (1u << 20)    // pad_enable
                | (7u << 22)    // pad_interval: 256 DWORDs (1KB)
                | (7u << 25));  // pad_amount: 8 DWORDs (32B)
  g1[1] = (int)(128u << 16);    // tensor_dim0 = 128 (lo16 in hi half)
  g1[2] = (int)(128u << 16);    // tensor_dim1 = 128
  g1[3] = (int)(16u << 16);     // tile_dim0 = 16
  g1[4] = (int)(16u | (128u << 16));  // tile_dim1 = 16, tile_dim2 = 128
  g1[5] = 128;                  // tensor_dim0_stride = 128 elements
  g1[6] = (int)(16384u << 16);  // tensor_dim1_stride[15:0] = 16384
  g1[7] = 0;                    // tensor_dim1_stride[47:16] = 0
  i32x4 g2;
  g2[0] = 128;                  // tensor_dim2 = 128
  g2[1] = 0; g2[2] = 0; g2[3] = 0;
  i32x4 g3;
  g3[0] = 0; g3[1] = 0; g3[2] = 0; g3[3] = 0;
  i32x8 g4;                     // extra group in 6-arg builtin form: zeros
  g4[0] = 0; g4[1] = 0; g4[2] = 0; g4[3] = 0;
  g4[4] = 0; g4[5] = 0; g4[6] = 0; g4[7] = 0;
  __builtin_amdgcn_tensor_load_to_lds(g0, g1, g2, g3, g4, 0);
}
#endif

__global__ void convert_weights(const float* __restrict__ wq,
                                const float* __restrict__ wk,
                                const float* __restrict__ wv,
                                h16* __restrict__ wq16, h16* __restrict__ wk16,
                                h16* __restrict__ wv16) {
  int idx = blockIdx.x * 256 + threadIdx.x;  // 65536 threads
  if (idx < 32 * 256) {
    wq16[idx] = (h16)wq[idx];
    wk16[idx] = (h16)wk[idx];
  }
  wv16[idx] = (h16)wv[idx];
}

__global__ __launch_bounds__(256) void
attn_win16_kernel(const float* __restrict__ x, const float* __restrict__ bq,
                  const float* __restrict__ bk, const float* __restrict__ bv,
                  const float* __restrict__ gamma,
                  const h16* __restrict__ wq16, const h16* __restrict__ wk16,
                  const h16* __restrict__ wv16, float* __restrict__ out,
                  float* __restrict__ att_out) {
  extern __shared__ __align__(16) char smem_raw[];
  h16* XT  = (h16*)smem_raw;          // xb^T [N][C] f16; later att [n][m]
  h16* Vs  = XT + 256 * XT_S;         // V [C][M] f16 (bv folded); TDM staging
  h16* Qs  = Vs + 256 * XT_S;         // Q [N][D] f16 (bq folded)
  h16* KTs = Qs + 256 * QK_S;         // K^T [M][D] f16 (bk folded)
  float* MASKN = (float*)(KTs + 256 * QK_S);  // 256 f32: 0 or -1e9

  const int tid  = threadIdx.x;
  const int lane = tid & 31;
  const int wave = tid >> 5;
  const int g  = blockIdx.x;
  const int b  = g >> 6, wi = (g >> 3) & 7, hi = g & 7;
  const size_t xbase =
      (size_t)b * (C_DIM * 128 * 128) + (size_t)(wi * 16) * 128 + hi * 16;

  const int arow = lane & 15;          // A row / B row within a tile
  const int abase = (lane >> 4) * 8;   // A k-offset within 32-chunk
  const int bsub  = (lane >> 4) * 8;   // B k-offset within 32-chunk

  // ---- Phase 1: window -> XT[n][c] (f16, transposed) ----
#if USE_TDM
  {
    const float* Xstage = (const float*)(smem_raw + VS_BYTE_OFF);
    for (int h = 0; h < 2; ++h) {
      if (wave == 0) {
        unsigned long long ga = (unsigned long long)(size_t)x +
                                4ull * (xbase + (size_t)h * 128 * 16384);
        tdm_load_window_f32(ga, (u32)VS_BYTE_OFF);
        __builtin_amdgcn_s_wait_tensorcnt(0);
      }
      __syncthreads();  // staged [c][i][j] f32 visible to all waves
      for (int c = 0; c < 128; ++c) {
        XT[tid * XT_S + h * 128 + c] = (h16)Xstage[c * XT_S + tid];
      }
      __syncthreads();  // staging consumed; safe to reuse / next TDM
    }
  }
#else
  {
    const int i = tid >> 4, j = tid & 15;
    for (int c = 0; c < C_DIM; ++c) {
      float v = x[xbase + (size_t)c * 16384 + i * 128 + j];
      XT[tid * XT_S + c] = (h16)v;
    }
    __syncthreads();
  }
#endif

  // ---- Phase 2: Q[n][d], KT[m][d] (biases folded). 64 tile jobs / 8 waves ----
  for (int jj = 0; jj < 8; ++jj) {
    const int job = wave * 8 + jj;
    const int which = job >> 5;        // 0 -> Q, 1 -> K
    const int dt = (job >> 4) & 1;     // d-tile (D=32 -> 2 tiles)
    const int nt = job & 15;           // n-tile
    const h16* Wg = (which ? wk16 : wq16) + (dt * 16 + arow) * C_DIM;
    v8h alo[8], ahi[8];
#pragma unroll
    for (int k = 0; k < 8; ++k) {
      const h16* p = Wg + k * 32 + abase;
      alo[k] = *(const v8h*)p;
      ahi[k] = *(const v8h*)(p + 16);
    }
    v8f acc = {};
    const h16* Bp = XT + (nt * 16 + arow) * XT_S + bsub;
#pragma unroll
    for (int k = 0; k < 8; ++k) {
      v8h blo = *(const v8h*)(Bp + k * 32);
      v8h bhi = *(const v8h*)(Bp + k * 32 + 16);
      acc = wmma_f16(mk16(alo[k], ahi[k]), mk16(blo, bhi), acc);
    }
    const float* bias = which ? bk : bq;
    h16* Dst = which ? KTs : Qs;
    const int n = nt * 16 + (lane & 15);
#pragma unroll
    for (int r = 0; r < 8; ++r) {
      const int d = dt * 16 + r + ((lane >> 4) * 8);
      Dst[n * QK_S + d] = (h16)(acc[r] + bias[d]);
    }
  }
  // padding mask per position m: column abs-sum of xb == row of XT
  {
    float s = 0.f;
    for (int c = 0; c < C_DIM; ++c) s += fabsf((float)XT[tid * XT_S + c]);
    MASKN[tid] = (s > 0.f) ? 0.f : -1e9f;
  }

  // ---- Phase 3: V[c][m] = Wv·xb + bv. 2 c-tiles per wave, A staged in regs ----
  for (int rep = 0; rep < 2; ++rep) {
    const int c0 = (wave * 2 + rep) * 16;
    const h16* Wg = wv16 + (c0 + arow) * C_DIM;
    __builtin_prefetch(Wg, 0, 3);  // global_prefetch_b8
    v8h alo[8], ahi[8];
#pragma unroll
    for (int k = 0; k < 8; ++k) {
      const h16* p = Wg + k * 32 + abase;
      alo[k] = *(const v8h*)p;
      ahi[k] = *(const v8h*)(p + 16);
    }
    float biasv[8];
#pragma unroll
    for (int r = 0; r < 8; ++r) biasv[r] = bv[c0 + r + ((lane >> 4) * 8)];
    for (int mt = 0; mt < 16; ++mt) {
      v8f acc = {};
      const h16* Bp = XT + (mt * 16 + arow) * XT_S + bsub;
#pragma unroll
      for (int k = 0; k < 8; ++k) {
        v8h blo = *(const v8h*)(Bp + k * 32);
        v8h bhi = *(const v8h*)(Bp + k * 32 + 16);
        acc = wmma_f16(mk16(alo[k], ahi[k]), mk16(blo, bhi), acc);
      }
      const int m = mt * 16 + (lane & 15);
#pragma unroll
      for (int r = 0; r < 8; ++r) {
        const int c = c0 + r + ((lane >> 4) * 8);
        Vs[c * XT_S + m] = (h16)(acc[r] + biasv[r]);
      }
    }
  }
  __syncthreads();  // XT reads done; Q/KT/V ready

  // ---- Phase 4: energy[n][m] = Q·K^T, one WMMA per tile (K=32). Overwrites XT ----
  for (int idx = 0; idx < 32; ++idx) {
    const int tj = wave * 32 + idx;
    const int nt = tj >> 4, mt = tj & 15;
    const h16* Ap = Qs + (nt * 16 + arow) * QK_S + abase;
    v8h alo = *(const v8h*)Ap;
    v8h ahi = *(const v8h*)(Ap + 16);
    const h16* Bp = KTs + (mt * 16 + arow) * QK_S + bsub;
    v8h blo = *(const v8h*)Bp;
    v8h bhi = *(const v8h*)(Bp + 16);
    v8f acc = {};
    acc = wmma_f16(mk16(alo, ahi), mk16(blo, bhi), acc);
    const int m = mt * 16 + (lane & 15);
#pragma unroll
    for (int r = 0; r < 8; ++r) {
      const int n = nt * 16 + r + ((lane >> 4) * 8);
      XT[n * XT_S + m] = (h16)acc[r];
    }
  }
  __syncthreads();

  // ---- Phase 5: softmax per row (thread t owns row t); emit f32 att output ----
  {
    const int n = tid;
    h16* row = XT + n * XT_S;
    float mx = -3.4e38f;
    for (int m = 0; m < N_DIM; ++m) {
      float e = (float)row[m] + MASKN[m];
      mx = fmaxf(mx, e);
    }
    float s = 0.f;
    for (int m = 0; m < N_DIM; ++m) {
      float e = __expf((float)row[m] + MASKN[m] - mx);
      s += e;
      row[m] = (h16)e;
    }
    const float inv = 1.f / s;
    float* arow_g = att_out + (size_t)g * 65536 + (size_t)n * 256;
    for (int m = 0; m < N_DIM; ++m) {
      float a = (float)row[m] * inv;
      row[m] = (h16)a;
      arow_g[m] = a;
    }
  }
  __syncthreads();

  // ---- Phase 6: out[c][n] = gamma * (V·att^T) + x ----
  const float gm = gamma[0];
  for (int rep = 0; rep < 2; ++rep) {
    const int c0 = (wave * 2 + rep) * 16;
    v8h alo[8], ahi[8];
#pragma unroll
    for (int k = 0; k < 8; ++k) {
      const h16* p = Vs + (c0 + arow) * XT_S + k * 32 + abase;
      alo[k] = *(const v8h*)p;
      ahi[k] = *(const v8h*)(p + 16);
    }
    for (int nt = 0; nt < 16; ++nt) {
      v8f acc = {};
      const h16* Bp = XT + (nt * 16 + arow) * XT_S + bsub;
#pragma unroll
      for (int k = 0; k < 8; ++k) {
        v8h blo = *(const v8h*)(Bp + k * 32);
        v8h bhi = *(const v8h*)(Bp + k * 32 + 16);
        acc = wmma_f16(mk16(alo[k], ahi[k]), mk16(blo, bhi), acc);
      }
      const int j = lane & 15;
#pragma unroll
      for (int r = 0; r < 8; ++r) {
        const int c = c0 + r + ((lane >> 4) * 8);
        const size_t gi = xbase + (size_t)c * 16384 + nt * 128 + j;
        out[gi] = gm * acc[r] + x[gi];
      }
    }
  }
}

extern "C" void kernel_launch(void* const* d_in, const int* in_sizes, int n_in,
                              void* d_out, int out_size, void* d_ws,
                              size_t ws_size, hipStream_t stream) {
  (void)in_sizes; (void)n_in; (void)out_size; (void)ws_size;
  const float* x     = (const float*)d_in[0];
  const float* Wq    = (const float*)d_in[1];
  const float* bq    = (const float*)d_in[2];
  const float* Wk    = (const float*)d_in[3];
  const float* bk    = (const float*)d_in[4];
  const float* Wv    = (const float*)d_in[5];
  const float* bv    = (const float*)d_in[6];
  const float* gamma = (const float*)d_in[7];

  h16* wq16 = (h16*)d_ws;             // 32*256 halves
  h16* wk16 = wq16 + 32 * 256;        // 32*256 halves
  h16* wv16 = wk16 + 32 * 256;        // 256*256 halves

  float* outp = (float*)d_out;
  float* attp = outp + (size_t)4 * 256 * 128 * 128;

  convert_weights<<<256, 256, 0, stream>>>(Wq, Wk, Wv, wq16, wk16, wv16);

  // LDS: 2*(256*264) + 2*(256*40) halves + 256 f32 = 312320 bytes
  const size_t lds_bytes = (size_t)(2 * 256 * XT_S + 2 * 256 * QK_S) * 2 + 256 * 4;
  attn_win16_kernel<<<256, 256, lds_bytes, stream>>>(
      x, bq, bk, bv, gamma, wq16, wk16, wv16, outp, attp);
}
